// MyConv2d_87600152969638
// MI455X (gfx1250) — matrix operand order repaired and checked by reading, not compiled, not run on hardware
//
#include <hip/hip_runtime.h>
#include <hip/hip_bf16.h>

typedef __attribute__((ext_vector_type(16))) __bf16          v16bf;
typedef __attribute__((ext_vector_type(8)))  float           v8f;
typedef __attribute__((ext_vector_type(16))) unsigned short  v16us;

#define CIN    128
#define COUT   256
#define HWDIM  56
#define NPIX   (HWDIM * HWDIM)   /* 3136 */
#define KW9    9

/* ---- workspace layout (bytes) ---- */
#define XSZ   ((size_t)32 * NPIX * CIN * 2)        /* 25,690,112 : one bf16 image set (NHWC) */
#define WSZ   ((size_t)COUT * KW9 * CIN * 2)       /*    589,824 : one bf16 weight set [k][tap][c] */
#define XHI_OFF  ((size_t)0)
#define XLO_OFF  (XSZ)
#define WHI_OFF  (2 * XSZ)
#define WLO_OFF  (2 * XSZ + WSZ)
#define Z_OFF    (2 * XSZ + 2 * WSZ)               /* 512 B zero page for padding redirect */
#define WS_NEED  (Z_OFF + 512)

struct __align__(32) Frag32B { uint4 a, b; };

// f32 -> bf16 hi (truncation) + bf16 lo (RNE of exact residual). hi+lo ~ 2^-17 rel.
__device__ __forceinline__ void split_bf16(float x, unsigned short& hi16, unsigned short& lo16) {
    unsigned int u  = __builtin_bit_cast(unsigned int, x);
    unsigned int uh = u & 0xFFFF0000u;
    float fl = x - __builtin_bit_cast(float, uh);
    unsigned int ul = __builtin_bit_cast(unsigned int, fl);
    ul += 0x7FFFu + ((ul >> 16) & 1u);
    hi16 = (unsigned short)(uh >> 16);
    lo16 = (unsigned short)(ul >> 16);
}

/* ================= preprocessing: x NCHW f32 -> NHWC bf16 hi/lo ================= */
__global__ __launch_bounds__(256) void split_x_kernel(const float* __restrict__ x, char* __restrict__ ws) {
    const int t = blockIdx.x * 256 + threadIdx.x;      // 0 .. 32*3136-1
    const int n = t / NPIX, p = t % NPIX;
    const float* xp = x + (size_t)n * CIN * NPIX + p;  // coalesced reads (contiguous p)
    unsigned short* hb = (unsigned short*)(ws + XHI_OFF) + (size_t)t * CIN;
    unsigned short* lb = (unsigned short*)(ws + XLO_OFF) + (size_t)t * CIN;
    #pragma unroll 4
    for (int c0 = 0; c0 < CIN; c0 += 8) {
        unsigned short h[8], l[8];
        #pragma unroll
        for (int i = 0; i < 8; ++i) split_bf16(xp[(size_t)(c0 + i) * NPIX], h[i], l[i]);
        uint4 uh, ul;
        uh.x = h[0] | (h[1] << 16); uh.y = h[2] | (h[3] << 16);
        uh.z = h[4] | (h[5] << 16); uh.w = h[6] | (h[7] << 16);
        ul.x = l[0] | (l[1] << 16); ul.y = l[2] | (l[3] << 16);
        ul.z = l[4] | (l[5] << 16); ul.w = l[6] | (l[7] << 16);
        *(uint4*)(hb + c0) = uh;
        *(uint4*)(lb + c0) = ul;
    }
}

/* ====== preprocessing: w OIHW f32 -> [k][tap][c] bf16 hi/lo; also zero page ====== */
__global__ __launch_bounds__(256) void split_w_kernel(const float* __restrict__ wgt, char* __restrict__ ws) {
    const int t = blockIdx.x * 256 + threadIdx.x;      // 0 .. 2303
    const int k = t / KW9, rs = t % KW9;
    const float* wp = wgt + (size_t)k * (CIN * KW9) + rs;   // element (k,c,rs): k*1152 + c*9 + rs
    unsigned short* hb = (unsigned short*)(ws + WHI_OFF) + (size_t)t * CIN;
    unsigned short* lb = (unsigned short*)(ws + WLO_OFF) + (size_t)t * CIN;
    #pragma unroll 4
    for (int c0 = 0; c0 < CIN; c0 += 8) {
        unsigned short h[8], l[8];
        #pragma unroll
        for (int i = 0; i < 8; ++i) split_bf16(wp[(size_t)(c0 + i) * KW9], h[i], l[i]);
        uint4 uh, ul;
        uh.x = h[0] | (h[1] << 16); uh.y = h[2] | (h[3] << 16);
        uh.z = h[4] | (h[5] << 16); uh.w = h[6] | (h[7] << 16);
        ul.x = l[0] | (l[1] << 16); ul.y = l[2] | (l[3] << 16);
        ul.z = l[4] | (l[5] << 16); ul.w = l[6] | (l[7] << 16);
        *(uint4*)(hb + c0) = uh;
        *(uint4*)(lb + c0) = ul;
    }
    if (blockIdx.x == 0 && threadIdx.x < 32)
        ((uint4*)(ws + Z_OFF))[threadIdx.x] = make_uint4(0u, 0u, 0u, 0u);
}

/* ============== main implicit-GEMM conv, bf16x3 WMMA, 32px x 32ch per wave ============== */
__global__ __launch_bounds__(256) void conv3x3_wmma_pre(
    const char* __restrict__ ws, const float* __restrict__ bias, float* __restrict__ out)
{
    const int lane   = threadIdx.x & 31;
    const int wave   = threadIdx.x >> 5;
    const int laneHi = lane >> 4;
    const int lm     = lane & 15;

    const int n  = blockIdx.x / (NPIX / 32);            // 98 tiles of 32 pixels per image
    const int m0 = (blockIdx.x % (NPIX / 32)) * 32;

    // Two M tiles: pixels p0 = m0+lm, p1 = m0+16+lm
    int offHi0[KW9], offLo0[KW9], offHi1[KW9], offLo1[KW9];
    #pragma unroll
    for (int g = 0; g < 2; ++g) {
        const int p  = m0 + g * 16 + lm;
        const int ph = p / HWDIM, pw = p % HWDIM;
        #pragma unroll
        for (int r = 0; r < 3; ++r) {
            #pragma unroll
            for (int s = 0; s < 3; ++s) {
                const int hh = ph + r - 1, ww = pw + s - 1;
                const bool v = (hh >= 0) && (hh < HWDIM) && (ww >= 0) && (ww < HWDIM);
                const int pix = (n * NPIX + hh * HWDIM + ww) * (CIN * 2);   // bytes
                const int oh = (v ? pix            : (int)Z_OFF) + 16 * laneHi;
                const int ol = (v ? pix + (int)XSZ : (int)Z_OFF) + 16 * laneHi;
                if (g == 0) { offHi0[r * 3 + s] = oh; offLo0[r * 3 + s] = ol; }
                else        { offHi1[r * 3 + s] = oh; offLo1[r * 3 + s] = ol; }
            }
        }
    }

    const int k0 = wave * 32 + lm;                       // N-tile 0 out-channel of this lane
    const int vB = k0 * (KW9 * CIN * 2) + 16 * laneHi;   // bytes into weight arrays
    const char* wh = ws + WHI_OFF;
    const char* wl = ws + WLO_OFF;

    v8f acc00, acc01, acc10, acc11;
    {
        const float b0 = bias[k0], b1 = bias[k0 + 16];
        #pragma unroll
        for (int j = 0; j < 8; ++j) { acc00[j] = b0; acc01[j] = b1; acc10[j] = b0; acc11[j] = b1; }
    }

    #pragma unroll
    for (int c0i = 0; c0i < 4; ++c0i) {                  // channel chunks of 32
        const int c0b = c0i * 64;                        // c0*2 bytes
        #pragma unroll
        for (int rs = 0; rs < KW9; ++rs) {
            // A fragments: two 16B runs each (K = {0..7,16..23}+8*laneHi)
            const char* pa0h = ws + offHi0[rs] + c0b;
            const char* pa0l = ws + offLo0[rs] + c0b;
            const char* pa1h = ws + offHi1[rs] + c0b;
            const char* pa1l = ws + offLo1[rs] + c0b;
            Frag32B f0h = { *(const uint4*)(pa0h), *(const uint4*)(pa0h + 32) };
            Frag32B f0l = { *(const uint4*)(pa0l), *(const uint4*)(pa0l + 32) };
            Frag32B f1h = { *(const uint4*)(pa1h), *(const uint4*)(pa1h + 32) };
            Frag32B f1l = { *(const uint4*)(pa1l), *(const uint4*)(pa1l + 32) };
            const v16bf A0h = __builtin_bit_cast(v16bf, f0h);
            const v16bf A0l = __builtin_bit_cast(v16bf, f0l);
            const v16bf A1h = __builtin_bit_cast(v16bf, f1h);
            const v16bf A1l = __builtin_bit_cast(v16bf, f1l);

            // B fragments: two 16B runs each (K = {0..7,16..23}+8*laneHi)
            const int wb = vB + rs * (CIN * 2) + c0b;
            Frag32B fb0h = { *(const uint4*)(wh + wb),      *(const uint4*)(wh + wb + 32) };
            Frag32B fb0l = { *(const uint4*)(wl + wb),      *(const uint4*)(wl + wb + 32) };
            const int wb1 = wb + 16 * (KW9 * CIN * 2);     // N-tile 1: k0+16
            Frag32B fb1h = { *(const uint4*)(wh + wb1),     *(const uint4*)(wh + wb1 + 32) };
            Frag32B fb1l = { *(const uint4*)(wl + wb1),     *(const uint4*)(wl + wb1 + 32) };
            const v16bf B0h = __builtin_bit_cast(v16bf, fb0h);
            const v16bf B0l = __builtin_bit_cast(v16bf, fb0l);
            const v16bf B1h = __builtin_bit_cast(v16bf, fb1h);
            const v16bf B1l = __builtin_bit_cast(v16bf, fb1l);

            // 12 WMMAs, interleaved across 4 independent accumulators (reuse distance 4)
            acc00 = __builtin_amdgcn_wmma_f32_16x16x32_bf16(false, A0h, false, B0h, (short)0, acc00, false, false);
            acc10 = __builtin_amdgcn_wmma_f32_16x16x32_bf16(false, A1h, false, B0h, (short)0, acc10, false, false);
            acc01 = __builtin_amdgcn_wmma_f32_16x16x32_bf16(false, A0h, false, B1h, (short)0, acc01, false, false);
            acc11 = __builtin_amdgcn_wmma_f32_16x16x32_bf16(false, A1h, false, B1h, (short)0, acc11, false, false);

            acc00 = __builtin_amdgcn_wmma_f32_16x16x32_bf16(false, A0h, false, B0l, (short)0, acc00, false, false);
            acc10 = __builtin_amdgcn_wmma_f32_16x16x32_bf16(false, A1h, false, B0l, (short)0, acc10, false, false);
            acc01 = __builtin_amdgcn_wmma_f32_16x16x32_bf16(false, A0h, false, B1l, (short)0, acc01, false, false);
            acc11 = __builtin_amdgcn_wmma_f32_16x16x32_bf16(false, A1h, false, B1l, (short)0, acc11, false, false);

            acc00 = __builtin_amdgcn_wmma_f32_16x16x32_bf16(false, A0l, false, B0h, (short)0, acc00, false, false);
            acc10 = __builtin_amdgcn_wmma_f32_16x16x32_bf16(false, A1l, false, B0h, (short)0, acc10, false, false);
            acc01 = __builtin_amdgcn_wmma_f32_16x16x32_bf16(false, A0l, false, B1h, (short)0, acc01, false, false);
            acc11 = __builtin_amdgcn_wmma_f32_16x16x32_bf16(false, A1l, false, B1h, (short)0, acc11, false, false);
        }
    }

    const int prow0 = m0 + 8 * laneHi;
    float* o00 = out + ((size_t)n * COUT + k0)      * NPIX + prow0;        // M-tile 0, N-tile 0
    float* o01 = out + ((size_t)n * COUT + k0 + 16) * NPIX + prow0;        // M-tile 0, N-tile 1
    float* o10 = o00 + 16;                                                 // M-tile 1, N-tile 0
    float* o11 = o01 + 16;                                                 // M-tile 1, N-tile 1
    *reinterpret_cast<float4*>(o00)     = make_float4(acc00[0], acc00[1], acc00[2], acc00[3]);
    *reinterpret_cast<float4*>(o00 + 4) = make_float4(acc00[4], acc00[5], acc00[6], acc00[7]);
    *reinterpret_cast<float4*>(o01)     = make_float4(acc01[0], acc01[1], acc01[2], acc01[3]);
    *reinterpret_cast<float4*>(o01 + 4) = make_float4(acc01[4], acc01[5], acc01[6], acc01[7]);
    *reinterpret_cast<float4*>(o10)     = make_float4(acc10[0], acc10[1], acc10[2], acc10[3]);
    *reinterpret_cast<float4*>(o10 + 4) = make_float4(acc10[4], acc10[5], acc10[6], acc10[7]);
    *reinterpret_cast<float4*>(o11)     = make_float4(acc11[0], acc11[1], acc11[2], acc11[3]);
    *reinterpret_cast<float4*>(o11 + 4) = make_float4(acc11[4], acc11[5], acc11[6], acc11[7]);
}

/* ============== fallback (no workspace): on-the-fly split ============== */
__global__ __launch_bounds__(256) void conv3x3_wmma_fallback(
    const float* __restrict__ x, const float* __restrict__ wgt,
    const float* __restrict__ bias, float* __restrict__ out)
{
    const int lane   = threadIdx.x & 31;
    const int wave   = threadIdx.x >> 5;
    const int laneHi = lane >> 4;
    const int lm     = lane & 15;
    const int n  = blockIdx.x / (NPIX / 16);
    const int m0 = (blockIdx.x % (NPIX / 16)) * 16;
    const int p  = m0 + lm;
    const int ph = p / HWDIM, pw = p % HWDIM;

    int offs[KW9]; unsigned int vmsk[KW9];
    #pragma unroll
    for (int r = 0; r < 3; ++r)
        #pragma unroll
        for (int s = 0; s < 3; ++s) {
            int hh = ph + r - 1, ww = pw + s - 1;
            bool v = (hh >= 0) && (hh < HWDIM) && (ww >= 0) && (ww < HWDIM);
            offs[r * 3 + s] = v ? (hh * HWDIM + ww) : 0;
            vmsk[r * 3 + s] = v ? 0xFFFFFFFFu : 0u;
        }

    const int kb = wave * 32, k0 = kb + lm, k1 = kb + 16 + lm;
    v8f acc0, acc1;
    { const float b0 = bias[k0], b1 = bias[k1];
      #pragma unroll
      for (int j = 0; j < 8; ++j) { acc0[j] = b0; acc1[j] = b1; } }

    const float* xn = x + (size_t)n * CIN * NPIX;
    const float* w0 = wgt + (size_t)k0 * (CIN * KW9);
    const float* w1 = wgt + (size_t)k1 * (CIN * KW9);

    for (int c0 = 0; c0 < CIN; c0 += 32) {
        const float* xc  = xn + (size_t)(c0 + 16 * laneHi) * NPIX;
        const float* wc0 = w0 + (size_t)(c0 + 16 * laneHi) * KW9;
        const float* wc1 = w1 + (size_t)(c0 + 16 * laneHi) * KW9;
        #pragma unroll
        for (int rs = 0; rs < KW9; ++rs) {
            const int off = offs[rs]; const unsigned int vm = vmsk[rs];
            v16us ah, al;
            #pragma unroll
            for (int e = 0; e < 16; ++e) {
                const int ko = e;
                float v = xc[(size_t)ko * NPIX + off];
                v = __builtin_bit_cast(float, __builtin_bit_cast(unsigned int, v) & vm);
                unsigned short h16, l16; split_bf16(v, h16, l16);
                ah[e] = h16; al[e] = l16;
            }
            const v16bf Ah = __builtin_bit_cast(v16bf, ah);
            const v16bf Al = __builtin_bit_cast(v16bf, al);
            v16us bh, bl;
            #pragma unroll
            for (int e = 0; e < 16; ++e) {
                unsigned short h16, l16; split_bf16(wc0[(size_t)e * KW9 + rs], h16, l16);
                bh[e] = h16; bl[e] = l16;
            }
            { const v16bf Bh = __builtin_bit_cast(v16bf, bh);
              const v16bf Bl = __builtin_bit_cast(v16bf, bl);
              acc0 = __builtin_amdgcn_wmma_f32_16x16x32_bf16(false, Ah, false, Bh, (short)0, acc0, false, false);
              acc0 = __builtin_amdgcn_wmma_f32_16x16x32_bf16(false, Ah, false, Bl, (short)0, acc0, false, false);
              acc0 = __builtin_amdgcn_wmma_f32_16x16x32_bf16(false, Al, false, Bh, (short)0, acc0, false, false); }
            #pragma unroll
            for (int e = 0; e < 16; ++e) {
                unsigned short h16, l16; split_bf16(wc1[(size_t)e * KW9 + rs], h16, l16);
                bh[e] = h16; bl[e] = l16;
            }
            { const v16bf Bh = __builtin_bit_cast(v16bf, bh);
              const v16bf Bl = __builtin_bit_cast(v16bf, bl);
              acc1 = __builtin_amdgcn_wmma_f32_16x16x32_bf16(false, Ah, false, Bh, (short)0, acc1, false, false);
              acc1 = __builtin_amdgcn_wmma_f32_16x16x32_bf16(false, Ah, false, Bl, (short)0, acc1, false, false);
              acc1 = __builtin_amdgcn_wmma_f32_16x16x32_bf16(false, Al, false, Bh, (short)0, acc1, false, false); }
        }
    }
    const int prow = m0 + 8 * laneHi;
    float* o0 = out + ((size_t)n * COUT + k0) * NPIX + prow;
    float* o1 = out + ((size_t)n * COUT + k1) * NPIX + prow;
    *reinterpret_cast<float4*>(o0)     = make_float4(acc0[0], acc0[1], acc0[2], acc0[3]);
    *reinterpret_cast<float4*>(o0 + 4) = make_float4(acc0[4], acc0[5], acc0[6], acc0[7]);
    *reinterpret_cast<float4*>(o1)     = make_float4(acc1[0], acc1[1], acc1[2], acc1[3]);
    *reinterpret_cast<float4*>(o1 + 4) = make_float4(acc1[4], acc1[5], acc1[6], acc1[7]);
}

extern "C" void kernel_launch(void* const* d_in, const int* in_sizes, int n_in,
                              void* d_out, int out_size, void* d_ws, size_t ws_size,
                              hipStream_t stream) {
    (void)in_sizes; (void)n_in; (void)out_size;
    const float* x    = (const float*)d_in[0];
    const float* wgt  = (const float*)d_in[1];
    const float* bias = (const float*)d_in[2];
    float*       out  = (float*)d_out;

    if (ws_size >= WS_NEED) {
        char* ws = (char*)d_ws;
        split_x_kernel<<<dim3((32 * NPIX) / 256), dim3(256), 0, stream>>>(x, ws);
        split_w_kernel<<<dim3((COUT * KW9) / 256), dim3(256), 0, stream>>>(wgt, ws);
        conv3x3_wmma_pre<<<dim3(32 * (NPIX / 32)), dim3(256), 0, stream>>>(ws, bias, out);
    } else {
        conv3x3_wmma_fallback<<<dim3(32 * (NPIX / 16)), dim3(256), 0, stream>>>(x, wgt, bias, out);
    }
}
